// BiLSTM_CNN_CRF_38001870635646
// MI455X (gfx1250) — matrix operand order repaired and checked by reading, not compiled, not run on hardware
//
#include <hip/hip_runtime.h>
#include <hip/hip_bf16.h>

// ---------------------------------------------------------------------------
// BiLSTM-CNN-CRF forward for MI455X (gfx1250, wave32, WMMA).
// Sizes fixed per reference: B=64 S=512 L=16 CD=30 FN=4 KW=3 WD=200 D=320
//                            H=256 (4H=1024) T=17 VOCAB=50000 CHARS=128
// ---------------------------------------------------------------------------

typedef __attribute__((ext_vector_type(16))) __bf16        v16bf;
typedef __attribute__((ext_vector_type(8)))  float         v8f;
typedef __attribute__((ext_vector_type(4)))  unsigned int  u32x4;

union BFrag { u32x4 u[2]; v16bf v; };

__device__ __forceinline__ v8f wmma_bf16(v16bf a, v16bf b, v8f c) {
  // D = A(16x32 bf16) * B(32x16 bf16) + C(16x16 f32)
  return __builtin_amdgcn_wmma_f32_16x16x32_bf16(
      /*neg_a=*/false, a, /*neg_b=*/false, b,
      /*c_mod=*/(short)0, c, /*reuse_a=*/false, /*reuse_b=*/false);
}

__device__ __forceinline__ float sigmoidf_(float x) { return 1.0f / (1.0f + __expf(-x)); }

// ---------------------------------------------------------------------------
// f32 -> bf16 conversion
// ---------------------------------------------------------------------------
__global__ void cvt_bf16_kernel(const float* __restrict__ in, __bf16* __restrict__ out, int n) {
  int i = blockIdx.x * 256 + threadIdx.x;
  if (i < n) out[i] = (__bf16)in[i];
}

// ---------------------------------------------------------------------------
// Fused word-embedding gather + char-embedding + grouped conv1d + maxpool.
// One 128-thread block per word (B*S = 32768 words). Output: zbf [32768][320] bf16.
// ---------------------------------------------------------------------------
__global__ __launch_bounds__(128) void fuse_embed_kernel(
    const int* __restrict__ tok, const int* __restrict__ ctok,
    const float* __restrict__ wemb, const float* __restrict__ cemb,
    const float* __restrict__ convw, const float* __restrict__ convb,
    __bf16* __restrict__ zbf) {
  __shared__ float ce[16][30];
  __shared__ int cid[16];
  const int wrd = blockIdx.x;          // b*512 + s
  const int tid = threadIdx.x;
  if (tid < 16) cid[tid] = ctok[wrd * 16 + tid];
  __syncthreads();
  for (int i = tid; i < 16 * 30; i += 128) {
    int p = i / 30, c = i % 30;
    ce[p][c] = cemb[cid[p] * 30 + c];
  }
  const int t = tok[wrd];
  __syncthreads();
  // word embedding -> z[0:200]
  for (int i = tid; i < 200; i += 128)
    zbf[(size_t)wrd * 320 + i] = (__bf16)wemb[(size_t)t * 200 + i];
  // grouped conv (groups=30, 4 filters/group, KW=3, valid) + max over 14 positions
  if (tid < 120) {
    const int oc = tid, g = oc >> 2;
    const float w0 = convw[oc * 3 + 0], w1 = convw[oc * 3 + 1], w2 = convw[oc * 3 + 2];
    const float bb = convb[oc];
    float mx = -1e30f;
    for (int p = 0; p < 14; ++p) {
      float v = ce[p][g] * w0 + ce[p + 1][g] * w1 + ce[p + 2][g] * w2 + bb;
      mx = fmaxf(mx, v);
    }
    zbf[(size_t)wrd * 320 + 200 + oc] = (__bf16)mx;
  }
}

// ---------------------------------------------------------------------------
// GEMM1: gx[dir] = z @ w_ih[dir]^T + b[dir].  M=32768, N=1024, K=320, bf16 WMMA.
// One wave per 32x32 macro-tile (2x2 register blocking: 4 WMMA per 4 fragment
// loads per K-step -> 2x the arithmetic intensity of 1 tile/wave).
// 8 waves/block; blockIdx.y = direction.
// Output layout: gx [dir][s][b][1024] bf16 (per-step LSTM slices contiguous).
// ---------------------------------------------------------------------------
__global__ __launch_bounds__(256) void gemm_gx_kernel(
    const __bf16* __restrict__ zbf,
    const __bf16* __restrict__ wihf, const __bf16* __restrict__ wihb,
    const float* __restrict__ bf, const float* __restrict__ bb,
    __bf16* __restrict__ gx) {
  const int dir  = blockIdx.y;
  const __bf16* __restrict__ wih  = dir ? wihb : wihf;
  const float*  __restrict__ bias = dir ? bb : bf;
  __bf16* __restrict__ gxo = gx + (size_t)dir * 512 * 64 * 1024;

  const int wave = threadIdx.x >> 5;
  const int lane = threadIdx.x & 31;
  const int m    = lane & 15;
  const int half = lane >> 4;
  const long tile = (long)blockIdx.x * 8 + wave;  // 0 .. 32767 macro-tiles
  const int  tn = (int)(tile & 31);               // 32 N macro-tiles (32 cols)
  const long tm = tile >> 5;                      // 1024 M macro-tiles (32 rows)
  const long row0 = tm * 32;
  const int  n0   = tn * 32;

  v8f acc00 = {0.f,0.f,0.f,0.f,0.f,0.f,0.f,0.f};
  v8f acc01 = acc00, acc10 = acc00, acc11 = acc00;
  for (int k0 = 0; k0 < 320; k0 += 32) {
    // A fragments: rows row0+m and row0+16+m;
    // per-lane K chunks [k0+half*8 .. +7] and [k0+16+half*8 .. +7]
    const __bf16* ap0 = zbf + (size_t)(row0 + m) * 320 + k0 + half * 8;
    const __bf16* ap1 = ap0 + (size_t)16 * 320;
    BFrag fa0, fa1;
    fa0.u[0] = *(const u32x4*)(ap0);
    fa0.u[1] = *(const u32x4*)(ap0 + 16);
    fa1.u[0] = *(const u32x4*)(ap1);
    fa1.u[1] = *(const u32x4*)(ap1 + 16);
    // B fragments: cols n0+m and n0+16+m; K chunks [k0+half*8 .. +7] and [k0+16+half*8 .. +7]
    const __bf16* bp0 = wih + (size_t)(n0 + m) * 320 + k0 + half * 8;
    const __bf16* bp1 = bp0 + (size_t)16 * 320;
    BFrag fb0, fb1;
    fb0.u[0] = *(const u32x4*)(bp0);
    fb0.u[1] = *(const u32x4*)(bp0 + 16);
    fb1.u[0] = *(const u32x4*)(bp1);
    fb1.u[1] = *(const u32x4*)(bp1 + 16);
    acc00 = wmma_bf16(fa0.v, fb0.v, acc00);
    acc01 = wmma_bf16(fa0.v, fb1.v, acc01);
    acc10 = wmma_bf16(fa1.v, fb0.v, acc10);
    acc11 = wmma_bf16(fa1.v, fb1.v, acc11);
  }
  const float bc0 = bias[n0 + m];
  const float bc1 = bias[n0 + 16 + m];
  #pragma unroll
  for (int v = 0; v < 8; ++v) {
    const long r0 = row0 + v + 8 * half;       // word index = b*512 + s
    const long r1 = r0 + 16;
    const long b0_ = r0 >> 9, s0_ = r0 & 511;
    const long b1_ = r1 >> 9, s1_ = r1 & 511;
    gxo[(size_t)s0_ * 64 * 1024 + (size_t)b0_ * 1024 + n0 + m]      = (__bf16)(acc00[v] + bc0);
    gxo[(size_t)s0_ * 64 * 1024 + (size_t)b0_ * 1024 + n0 + 16 + m] = (__bf16)(acc01[v] + bc1);
    gxo[(size_t)s1_ * 64 * 1024 + (size_t)b1_ * 1024 + n0 + m]      = (__bf16)(acc10[v] + bc0);
    gxo[(size_t)s1_ * 64 * 1024 + (size_t)b1_ * 1024 + n0 + 16 + m] = (__bf16)(acc11[v] + bc1);
  }
}

// ---------------------------------------------------------------------------
// BiLSTM scan. grid (4, 2): blockIdx.x = 16-row batch slice, blockIdx.y = dir.
// 1024 threads = 32 waves; wave w computes N-tiles {2w, 2w+1} of the 16x1024
// recurrent product h @ w_hh^T each step with WMMA (A = h in LDS, B = w_hh in
// L2; w_hh is 512 KB bf16 -> L2-resident under the 192 MB L2).
// gx[t] is added in the elementwise phase with fully coalesced bf16 loads
// (consecutive threads -> consecutive columns), so the WMMA accumulator
// starts at zero and no strided C-fragment gather is needed.
// LDS: h bf16 16x256 (8KB) + c f32 16x256 (16KB) + gates bf16 16x1024 (32KB).
// ---------------------------------------------------------------------------
__global__ __launch_bounds__(1024) void lstm_kernel(
    const __bf16* __restrict__ gx,
    const __bf16* __restrict__ whhf, const __bf16* __restrict__ whhb,
    float* __restrict__ rnn /* [B][S][512] */) {
  __shared__ __bf16 hbuf[16 * 256];
  __shared__ float  cbuf[16 * 256];
  __shared__ __bf16 gates[16 * 1024];

  const int dir = blockIdx.y;
  const int b0  = blockIdx.x * 16;
  const __bf16* __restrict__ whh = dir ? whhb : whhf;
  const __bf16* __restrict__ gxd = gx + (size_t)dir * 512 * 64 * 1024;

  for (int i = threadIdx.x; i < 16 * 256; i += 1024) {
    hbuf[i] = (__bf16)0.0f;
    cbuf[i] = 0.0f;
  }
  __syncthreads();

  const int wave = threadIdx.x >> 5;
  const int lane = threadIdx.x & 31;
  const int m    = lane & 15;
  const int half = lane >> 4;

  for (int t = 0; t < 512; ++t) {
    const int s = dir ? (511 - t) : t;
    const __bf16* __restrict__ gxs = gxd + (size_t)s * 64 * 1024 + (size_t)b0 * 1024;
    // ---- gates_hw = h @ w_hh^T  (2 N-tiles per wave, 8 K-steps) ----
    #pragma unroll
    for (int i = 0; i < 2; ++i) {
      const int n0 = (wave * 2 + i) * 16;
      v8f acc = {0.f,0.f,0.f,0.f,0.f,0.f,0.f,0.f};
      for (int k0 = 0; k0 < 256; k0 += 32) {
        const __bf16* ap = hbuf + m * 256 + k0 + half * 8;
        BFrag fa;
        fa.u[0] = *(const u32x4*)(ap);
        fa.u[1] = *(const u32x4*)(ap + 16);
        const __bf16* bp = whh + (size_t)(n0 + m) * 256 + k0 + half * 8;
        BFrag fb;
        fb.u[0] = *(const u32x4*)(bp);
        fb.u[1] = *(const u32x4*)(bp + 16);
        acc = wmma_bf16(fa.v, fb.v, acc);
      }
      #pragma unroll
      for (int v = 0; v < 8; ++v)
        gates[(v + 8 * half) * 1024 + n0 + m] = (__bf16)acc[v];
    }
    __syncthreads();
    // ---- elementwise cell update (torch gate order i,f,g,o);
    //      gx added here with coalesced loads ----
    for (int u = threadIdx.x; u < 16 * 256; u += 1024) {
      const int mm = u >> 8, jj = u & 255;
      const __bf16* gxr = gxs + (size_t)mm * 1024;
      const float gi = (float)gates[mm * 1024 + jj]       + (float)gxr[jj];
      const float gf = (float)gates[mm * 1024 + 256 + jj] + (float)gxr[256 + jj];
      const float gg = (float)gates[mm * 1024 + 512 + jj] + (float)gxr[512 + jj];
      const float go = (float)gates[mm * 1024 + 768 + jj] + (float)gxr[768 + jj];
      float c = cbuf[u];
      c = sigmoidf_(gf) * c + sigmoidf_(gi) * tanhf(gg);
      const float h = sigmoidf_(go) * tanhf(c);
      cbuf[u] = c;
      hbuf[u] = (__bf16)h;
      rnn[((size_t)(b0 + mm) * 512 + s) * 512 + dir * 256 + jj] = h;
    }
    __syncthreads();
  }
}

// ---------------------------------------------------------------------------
// Emissions: [B*S, 512] @ cls_w^T[512,17] + cls_b.  N=17 is too small for WMMA;
// plain f32 dot products (0.57 GFLOP total, irrelevant).
// ---------------------------------------------------------------------------
__global__ void emissions_kernel(const float* __restrict__ rnn,
                                 const float* __restrict__ clsw,
                                 const float* __restrict__ clsb,
                                 float* __restrict__ emis) {
  const long idx = (long)blockIdx.x * 256 + threadIdx.x;
  if (idx >= 32768L * 17) return;
  const int  cls = (int)(idx % 17);
  const long w   = idx / 17;  // b*512 + s
  const float* r  = rnn + w * 512;
  const float* cw = clsw + cls * 512;
  float acc = clsb[cls];
  for (int k = 0; k < 512; ++k) acc += r[k] * cw[k];
  emis[w * 17 + cls] = acc;
}

// ---------------------------------------------------------------------------
// CRF forward algorithm (log partition). Single block, 1024 threads,
// (b,j) state pairs in LDS, double-buffered alpha, 511 sequential steps.
// ---------------------------------------------------------------------------
__global__ __launch_bounds__(1024) void crf_forward_kernel(
    const float* __restrict__ emis, const int* __restrict__ mask,
    const float* __restrict__ start_t, const float* __restrict__ end_t,
    const float* __restrict__ trans, float* __restrict__ denom) {
  __shared__ float alpha[2][64][17];
  __shared__ float tr[17][17];
  const int tid = threadIdx.x;
  if (tid < 289) tr[tid / 17][tid % 17] = trans[tid];
  for (int p = tid; p < 64 * 17; p += 1024) {
    const int b = p / 17, j = p % 17;
    alpha[0][b][j] = start_t[j] + emis[(size_t)b * 512 * 17 + j];
  }
  __syncthreads();
  int cur = 0;
  for (int t = 1; t < 512; ++t) {
    for (int p = tid; p < 64 * 17; p += 1024) {
      const int b = p / 17, j = p % 17;
      float mx = -1e30f;
      for (int i = 0; i < 17; ++i) mx = fmaxf(mx, alpha[cur][b][i] + tr[i][j]);
      float sm = 0.f;
      for (int i = 0; i < 17; ++i) sm += __expf(alpha[cur][b][i] + tr[i][j] - mx);
      const float nxt = mx + __logf(sm) + emis[((size_t)b * 512 + t) * 17 + j];
      alpha[cur ^ 1][b][j] = (mask[b * 512 + t] > 0) ? nxt : alpha[cur][b][j];
    }
    __syncthreads();
    cur ^= 1;
  }
  if (tid < 64) {
    float mx = -1e30f;
    for (int j = 0; j < 17; ++j) mx = fmaxf(mx, alpha[cur][tid][j] + end_t[j]);
    float sm = 0.f;
    for (int j = 0; j < 17; ++j) sm += __expf(alpha[cur][tid][j] + end_t[j] - mx);
    denom[tid] = mx + __logf(sm);
  }
}

// ---------------------------------------------------------------------------
// Gold-path score + final loss = -mean(num - denom).
// ---------------------------------------------------------------------------
__global__ __launch_bounds__(64) void crf_loss_final_kernel(
    const float* __restrict__ emis, const int* __restrict__ tags,
    const int* __restrict__ mask, const float* __restrict__ start_t,
    const float* __restrict__ end_t, const float* __restrict__ trans,
    const float* __restrict__ denom, float* __restrict__ out_loss) {
  __shared__ float red[64];
  const int b = threadIdx.x;
  const int* tg = tags + b * 512;
  const int* mk = mask + b * 512;
  float num = start_t[tg[0]];
  int cnt = 0;
  for (int t = 0; t < 512; ++t) {
    if (mk[t] > 0) {
      num += emis[((size_t)b * 512 + t) * 17 + tg[t]];
      ++cnt;
    }
  }
  for (int t = 1; t < 512; ++t)
    if (mk[t] > 0) num += trans[tg[t - 1] * 17 + tg[t]];
  if (cnt < 1) cnt = 1;
  num += end_t[tg[cnt - 1]];
  red[b] = num - denom[b];
  __syncthreads();
  if (b == 0) {
    float s = 0.f;
    for (int i = 0; i < 64; ++i) s += red[i];
    out_loss[0] = -s / 64.0f;
  }
}

// ---------------------------------------------------------------------------
// Viterbi decode: forward max-product with backpointers (u8 in ws), then
// per-batch backtrace; tags written as float to d_out[1..].
// ---------------------------------------------------------------------------
__global__ __launch_bounds__(1024) void crf_viterbi_kernel(
    const float* __restrict__ emis, const int* __restrict__ mask,
    const float* __restrict__ start_t, const float* __restrict__ end_t,
    const float* __restrict__ trans, unsigned char* __restrict__ bps,
    float* __restrict__ out_tokens) {
  __shared__ float alpha[2][64][17];
  __shared__ float tr[17][17];
  const int tid = threadIdx.x;
  if (tid < 289) tr[tid / 17][tid % 17] = trans[tid];
  for (int p = tid; p < 64 * 17; p += 1024) {
    const int b = p / 17, j = p % 17;
    alpha[0][b][j] = start_t[j] + emis[(size_t)b * 512 * 17 + j];
  }
  __syncthreads();
  int cur = 0;
  for (int t = 1; t < 512; ++t) {
    for (int p = tid; p < 64 * 17; p += 1024) {
      const int b = p / 17, j = p % 17;
      float best = -1e30f;
      int bi = 0;
      for (int i = 0; i < 17; ++i) {
        const float v = alpha[cur][b][i] + tr[i][j];
        if (v > best) { best = v; bi = i; }
      }
      const bool on = mask[b * 512 + t] > 0;
      alpha[cur ^ 1][b][j] =
          on ? (best + emis[((size_t)b * 512 + t) * 17 + j]) : alpha[cur][b][j];
      bps[((size_t)(t - 1) * 64 + b) * 17 + j] = (unsigned char)(on ? bi : j);
    }
    __syncthreads();
    cur ^= 1;
  }
  if (tid < 64) {
    const int b = tid;
    float best = -1e30f;
    int tag = 0;
    for (int j = 0; j < 17; ++j) {
      const float v = alpha[cur][b][j] + end_t[j];
      if (v > best) { best = v; tag = j; }
    }
    out_tokens[b * 512 + 511] = (float)tag;
    for (int t = 510; t >= 0; --t) {
      tag = bps[((size_t)t * 64 + b) * 17 + tag];
      out_tokens[b * 512 + t] = (float)tag;
    }
  }
}

// ---------------------------------------------------------------------------
// Host orchestration
// ---------------------------------------------------------------------------
extern "C" void kernel_launch(void* const* d_in, const int* in_sizes, int n_in,
                              void* d_out, int out_size, void* d_ws, size_t ws_size,
                              hipStream_t stream) {
  (void)in_sizes; (void)n_in; (void)out_size; (void)ws_size;
  const int*   token_ids = (const int*)d_in[0];
  const int*   ctok      = (const int*)d_in[1];
  const int*   labels    = (const int*)d_in[2];
  const int*   mask      = (const int*)d_in[3];
  const float* wemb      = (const float*)d_in[4];
  const float* cemb      = (const float*)d_in[5];
  const float* convw     = (const float*)d_in[6];
  const float* convb     = (const float*)d_in[7];
  const float* w_ih_f    = (const float*)d_in[8];
  const float* w_hh_f    = (const float*)d_in[9];
  const float* b_f       = (const float*)d_in[10];
  const float* w_ih_b    = (const float*)d_in[11];
  const float* w_hh_b    = (const float*)d_in[12];
  const float* b_b       = (const float*)d_in[13];
  const float* cls_w     = (const float*)d_in[14];
  const float* cls_b     = (const float*)d_in[15];
  const float* start_t   = (const float*)d_in[16];
  const float* end_t     = (const float*)d_in[17];
  const float* trans     = (const float*)d_in[18];
  float* out = (float*)d_out;

  // workspace carve-up (256B aligned)
  char* ws = (char*)d_ws;
  size_t off = 0;
  auto take = [&](size_t bytes) -> char* {
    char* p = ws + off;
    off = (off + bytes + 255) & ~(size_t)255;
    return p;
  };
  __bf16* zbf     = (__bf16*)take(32768UL * 320 * 2);        // 20 MB
  __bf16* wihf_bf = (__bf16*)take(1024UL * 320 * 2);
  __bf16* wihb_bf = (__bf16*)take(1024UL * 320 * 2);
  __bf16* whhf_bf = (__bf16*)take(1024UL * 256 * 2);
  __bf16* whhb_bf = (__bf16*)take(1024UL * 256 * 2);
  __bf16* gx      = (__bf16*)take(2UL * 512 * 64 * 1024 * 2); // 134 MB
  float*  rnn     = (float*)take(64UL * 512 * 512 * 4);       // 64 MB
  float*  emis    = (float*)take(64UL * 512 * 17 * 4);        // 2.2 MB
  float*  denom   = (float*)take(64 * 4);
  unsigned char* bps = (unsigned char*)take(511UL * 64 * 17); // 0.56 MB

  // 1) weights -> bf16
  cvt_bf16_kernel<<<(327680 + 255) / 256, 256, 0, stream>>>(w_ih_f, wihf_bf, 327680);
  cvt_bf16_kernel<<<(327680 + 255) / 256, 256, 0, stream>>>(w_ih_b, wihb_bf, 327680);
  cvt_bf16_kernel<<<(262144 + 255) / 256, 256, 0, stream>>>(w_hh_f, whhf_bf, 262144);
  cvt_bf16_kernel<<<(262144 + 255) / 256, 256, 0, stream>>>(w_hh_b, whhb_bf, 262144);

  // 2) embeddings + char CNN -> z (bf16)
  fuse_embed_kernel<<<32768, 128, 0, stream>>>(token_ids, ctok, wemb, cemb, convw, convb, zbf);

  // 3) gx[dir] = z @ w_ih^T + b  (bf16 WMMA, 32768 32x32 macro-tiles per dir)
  gemm_gx_kernel<<<dim3(4096, 2), 256, 0, stream>>>(zbf, wihf_bf, wihb_bf, b_f, b_b, gx);

  // 4) BiLSTM scan (WMMA recurrent matmul, state in LDS, 32 waves/WG)
  lstm_kernel<<<dim3(4, 2), 1024, 0, stream>>>(gx, whhf_bf, whhb_bf, rnn);

  // 5) emissions
  emissions_kernel<<<(32768 * 17 + 255) / 256, 256, 0, stream>>>(rnn, cls_w, cls_b, emis);

  // 6) CRF loss
  crf_forward_kernel<<<1, 1024, 0, stream>>>(emis, mask, start_t, end_t, trans, denom);
  crf_loss_final_kernel<<<1, 64, 0, stream>>>(emis, labels, mask, start_t, end_t, trans, denom, out);

  // 7) Viterbi decode -> d_out[1..]
  crf_viterbi_kernel<<<1, 1024, 0, stream>>>(emis, mask, start_t, end_t, trans, bps, out + 1);
}
